// PowerSpectrumModel_18889266168220
// MI455X (gfx1250) — matrix operand order repaired and checked by reading, not compiled, not run on hardware
//
#include <hip/hip_runtime.h>
#include <hip/hip_bf16.h>

// ---------------------------------------------------------------------------
// Problem constants (match the reference)
// ---------------------------------------------------------------------------
#define NATOMS  8192
#define NBATCH  64
#define NEDGES  262144
#define NSPEC   4
#define NMAXR   8
#define RCUT    5.0f
#define PSDIM   4096     // (4*8)^2 * 4
#define HID1    1024
#define HID2    1024
#define PI_F    3.14159265358979323846f

typedef __attribute__((ext_vector_type(16))) _Float16 v16h;
typedef __attribute__((ext_vector_type(8)))  _Float16 h8;
typedef __attribute__((ext_vector_type(8)))  float    v8f;
typedef __attribute__((ext_vector_type(4)))  unsigned int u32x4;
typedef __attribute__((ext_vector_type(8)))  int          i32x8;
typedef __attribute__((ext_vector_type(4)))  int          i32x4;

#if defined(__has_builtin)
#if __has_builtin(__builtin_amdgcn_tensor_load_to_lds) && __has_builtin(__builtin_amdgcn_s_wait_tensorcnt)
#define USE_TDM 1
#endif
#endif

// ---------------------------------------------------------------------------
// Utility kernels
// ---------------------------------------------------------------------------
__global__ void zero_f32_kernel(float* __restrict__ p, int n) {
    int t = blockIdx.x * blockDim.x + threadIdx.x;
    if (t < n) p[t] = 0.0f;
}

__global__ void cvt_f32_to_f16_kernel(const float* __restrict__ src,
                                      _Float16* __restrict__ dst, int n) {
    int t = blockIdx.x * blockDim.x + threadIdx.x;
    if (t < n) dst[t] = (_Float16)src[t];
}

// ---------------------------------------------------------------------------
// K1: edge featurization + segment scatter into c[N*S, 128] (f32 atomics, L2)
// ---------------------------------------------------------------------------
__global__ void edge_kernel(const float* __restrict__ pos,
                            const float* __restrict__ cells,
                            const int*   __restrict__ numbers,
                            const int*   __restrict__ eidx,
                            const float* __restrict__ eoff,
                            const int*   __restrict__ batch,
                            float* __restrict__ c) {
    int e = blockIdx.x * blockDim.x + threadIdx.x;
    if (e >= NEDGES) return;
    int i = eidx[e];
    int j = eidx[NEDGES + e];
    int bi = batch[i];
    const float* Mx = cells + (size_t)bi * 9;
    float o0 = eoff[e * 3 + 0], o1 = eoff[e * 3 + 1], o2 = eoff[e * 3 + 2];
    float sx = o0 * Mx[0] + o1 * Mx[3] + o2 * Mx[6];
    float sy = o0 * Mx[1] + o1 * Mx[4] + o2 * Mx[7];
    float sz = o0 * Mx[2] + o1 * Mx[5] + o2 * Mx[8];
    float rx = pos[j * 3 + 0] - pos[i * 3 + 0] + sx;
    float ry = pos[j * 3 + 1] - pos[i * 3 + 1] + sy;
    float rz = pos[j * 3 + 2] - pos[i * 3 + 2] + sz;
    float r  = sqrtf(rx * rx + ry * ry + rz * rz + 1e-12f);
    float inv = 1.0f / r;
    float x = rx * inv, y = ry * inv, z = rz * inv;
    float fc = (r < RCUT) ? 0.5f * (__cosf(PI_F * r / RCUT) + 1.0f) : 0.0f;

    float R[NMAXR];
    float w = PI_F * r / RCUT;
#pragma unroll
    for (int n = 0; n < NMAXR; ++n)
        R[n] = fc * __sinf(w * (float)(n + 1)) * inv;

    float x2 = x * x, y2 = y * y, z2 = z * z;
    float Y[16];
    Y[0]  = 0.28209479f;
    Y[1]  = 0.48860251f * y;
    Y[2]  = 0.48860251f * z;
    Y[3]  = 0.48860251f * x;
    Y[4]  = 1.09254843f * x * y;
    Y[5]  = 1.09254843f * y * z;
    Y[6]  = 0.31539157f * (3.0f * z2 - 1.0f);
    Y[7]  = 1.09254843f * x * z;
    Y[8]  = 0.54627422f * (x2 - y2);
    Y[9]  = 0.59004359f * y * (3.0f * x2 - y2);
    Y[10] = 2.89061144f * x * y * z;
    Y[11] = 0.45704579f * y * (5.0f * z2 - 1.0f);
    Y[12] = 0.37317633f * z * (5.0f * z2 - 3.0f);
    Y[13] = 0.45704579f * x * (5.0f * z2 - 1.0f);
    Y[14] = 1.44530572f * z * (x2 - y2);
    Y[15] = 0.59004359f * x * (x2 - 3.0f * y2);

    int sp = numbers[j];
    float* base = c + ((size_t)i * NSPEC + sp) * (NMAXR * 16);
    for (int n = 0; n < NMAXR; ++n) {
        float rn = R[n];
#pragma unroll
        for (int m = 0; m < 16; ++m)
            atomicAdd(base + n * 16 + m, rn * Y[m]);
    }
}

// ---------------------------------------------------------------------------
// K2: per-atom power spectrum + LayerNorm + W_ps dot. psn written as f16.
// One 256-thread block per atom.
// ---------------------------------------------------------------------------
__global__ void ps_layernorm_kernel(const float* __restrict__ c,
                                    const float* __restrict__ gamma,
                                    const float* __restrict__ beta,
                                    const float* __restrict__ Wps,
                                    _Float16* __restrict__ psn,
                                    float* __restrict__ psl) {
    int atom = blockIdx.x;
    int tid  = threadIdx.x;

    __shared__ float cA[512];            // c[atom] as 32 x 16
    __shared__ float ps[PSDIM];          // 16 KB
    __shared__ float redA[256];
    __shared__ float redB[256];
    __shared__ float s_mu, s_rstd;

    const float* csrc = c + (size_t)atom * 512;
    for (int t = tid; t < 512; t += 256) cA[t] = csrc[t];
    __syncthreads();

    const int sb[4] = {0, 1, 4, 9};
    const int eb[4] = {1, 4, 9, 16};

    float sum = 0.0f, sum2 = 0.0f;
#pragma unroll
    for (int it = 0; it < 16; ++it) {
        int idx = tid + it * 256;        // idx = x*128 + y*4 + l
        int xI = idx >> 7;
        int yI = (idx >> 2) & 31;
        int l  = idx & 3;
        float p = 0.0f;
        for (int m = sb[l]; m < eb[l]; ++m)
            p += cA[xI * 16 + m] * cA[yI * 16 + m];
        ps[idx] = p;
        sum  += p;
        sum2 += p * p;
    }
    redA[tid] = sum; redB[tid] = sum2;
    __syncthreads();
    for (int off = 128; off > 0; off >>= 1) {
        if (tid < off) { redA[tid] += redA[tid + off]; redB[tid] += redB[tid + off]; }
        __syncthreads();
    }
    if (tid == 0) {
        float mu  = redA[0] * (1.0f / PSDIM);
        float var = redB[0] * (1.0f / PSDIM) - mu * mu;   // population variance
        s_mu   = mu;
        s_rstd = rsqrtf(var + 1e-5f);
    }
    __syncthreads();
    float mu = s_mu, rstd = s_rstd;

    float pacc = 0.0f;
    _Float16* pdst = psn + (size_t)atom * PSDIM;
#pragma unroll
    for (int it = 0; it < 16; ++it) {
        int idx = tid + it * 256;
        float v = (ps[idx] - mu) * rstd * gamma[idx] + beta[idx];
        pdst[idx] = (_Float16)v;
        pacc += v * Wps[idx];
    }
    __syncthreads();
    redA[tid] = pacc;
    __syncthreads();
    for (int off = 128; off > 0; off >>= 1) {
        if (tid < off) redA[tid] += redA[tid + off];
        __syncthreads();
    }
    if (tid == 0) psl[atom] = redA[0];
}

// ---------------------------------------------------------------------------
// K3: WMMA GEMM + SiLU.   C(MxN,f16) = silu(A(MxK,f16) @ B(KxN,f16))
// Block = 256 threads = 8 waves. BM=128, BN=128, BK=32.
// Wave wid: wr0=(wid&3)*32, wc0=(wid>>2)*64 -> 32x64 macro-tile = 2x4 WMMA.
// A tile staged Global->LDS by the Tensor Data Mover (padded 40-half rows via
// the D# pad fields); B staged transposed (N-major) so fragment reads are
// contiguous 16B ds_load chunks.
// ---------------------------------------------------------------------------
#define BM 128
#define BN 128
#define BK 32
#define BKP 40   // padded LDS K-stride in halves (16B aligned, conflict-free)

__global__ __launch_bounds__(256)
void gemm_silu_kernel(const _Float16* __restrict__ A,
                      const _Float16* __restrict__ B,
                      _Float16* __restrict__ C,
                      int M, int Nn, int K) {
    __shared__ _Float16 As[BM * BKP];   // 10 KB
    __shared__ _Float16 Bs[BN * BKP];   // 10 KB (transposed: [n][k])

    int tid  = threadIdx.x;
    int lane = tid & 31;
    int wid  = tid >> 5;
    int wr0  = (wid & 3) * 32;          // wave row base within block tile
    int wc0  = (wid >> 2) * 64;         // wave col base within block tile
    int mrow = blockIdx.y * BM;
    int ncol = blockIdx.x * BN;

    int rsub = lane & 15;
    int hi   = lane >> 4;               // 0 or 1 (lane half)
    int ksel = hi * 8;                  // unused: A fragments also read K at kbase
    int kbase = hi * 16;                // A/B-fragment K base: contiguous k = 16*hi + s

    v8f acc[2][4];
#pragma unroll
    for (int i = 0; i < 2; ++i)
#pragma unroll
        for (int j = 0; j < 4; ++j)
            acc[i][j] = (v8f){0.f, 0.f, 0.f, 0.f, 0.f, 0.f, 0.f, 0.f};

    for (int k0 = 0; k0 < K; k0 += BK) {
        // ---- stage A tile 128x32 halves ----
#if defined(USE_TDM)
        if (wid == 0) {
            // Tensor DMA descriptor (§8.3/§8.4): 2-D tensor, tile 32 x 128,
            // data_size=2B, LDS pad: 4 dwords after every 16 dwords -> row
            // stride 40 halves, matching As[row*BKP + k].
            unsigned ldsA = (unsigned)(uintptr_t)(&As[0]);
            unsigned long long ga = (unsigned long long)(uintptr_t)A +
                ((unsigned long long)mrow * (unsigned long long)K +
                 (unsigned long long)k0) * 2ull;
            u32x4 g0;
            g0.x = 1u;                                        // count=1 (valid)
            g0.y = ldsA;                                      // lds_addr
            g0.z = (unsigned)(ga & 0xFFFFFFFFull);            // global_addr lo
            g0.w = (unsigned)((ga >> 32) & 0x01FFFFFFull)     // global_addr hi
                 | (2u << 30);                                // type = 2 (image)
            i32x8 g1;
            g1[0] = (int)((1u << 16)      // data_size = 2 bytes
                        | (1u << 20)      // pad_enable
                        | (3u << 22)      // pad_interval: 16 dwords
                        | (3u << 25));    // pad_amount:   4 dwords
            g1[1] = (int)(((unsigned)K & 0xFFFFu) << 16);                 // tensor_dim0 lo
            g1[2] = (int)(((unsigned)K >> 16) | (((unsigned)M & 0xFFFFu) << 16)); // dim0 hi | dim1 lo
            g1[3] = (int)(((unsigned)M >> 16) | (32u << 16));             // dim1 hi | tile_dim0=32
            g1[4] = (int)(unsigned)BM;                                    // tile_dim1 = 128
            g1[5] = (int)K;                                               // tensor_dim0_stride lo
            g1[6] = 0;                                                    // stride hi | dim1_stride lo
            g1[7] = 0;
            i32x4 gz4 = (i32x4){0, 0, 0, 0};
            i32x8 gz8 = (i32x8){0, 0, 0, 0, 0, 0, 0, 0};
            __builtin_amdgcn_tensor_load_to_lds(g0, g1, gz4, gz4, gz8, 0);
            __builtin_amdgcn_s_wait_tensorcnt(0);
        }
#else
#pragma unroll
        for (int it = 0; it < 2; ++it) {
            int chunk = tid + it * 256;          // 0..511
            int row   = chunk >> 2;              // 0..127
            int koff  = (chunk & 3) * 8;         // 0,8,16,24
            h8 v = *(const h8*)(A + (size_t)(mrow + row) * K + k0 + koff);
            *(h8*)(&As[row * BKP + koff]) = v;
        }
#endif
        // ---- stage B tile transposed: read 32 x 128, write Bs[n][k] ----
#pragma unroll
        for (int it = 0; it < 2; ++it) {
            int chunk = tid + it * 256;          // 0..511
            int kk    = chunk >> 4;              // 0..31
            int noff  = (chunk & 15) * 8;        // 0..120
            h8 v = *(const h8*)(B + (size_t)(k0 + kk) * Nn + ncol + noff);
#pragma unroll
            for (int h = 0; h < 8; ++h)
                Bs[(noff + h) * BKP + kk] = v[h];
        }
        // prefetch next A tile into caches while we compute
        if (k0 + BK < K)
            __builtin_prefetch(A + (size_t)(mrow + (tid >> 1)) * K + k0 + BK, 0, 3);
        __syncthreads();

        // ---- build fragments ----
        v16h afr[2], bfr[4];
#pragma unroll
        for (int t = 0; t < 2; ++t) {
            int row = wr0 + t * 16 + rsub;
            h8 lo = *(const h8*)(&As[row * BKP + kbase]);       // K = kbase..kbase+7
            h8 hp = *(const h8*)(&As[row * BKP + kbase + 8]);   // K = kbase+8..+15
#pragma unroll
            for (int q = 0; q < 8; ++q) { afr[t][q] = lo[q]; afr[t][q + 8] = hp[q]; }
        }
#pragma unroll
        for (int t = 0; t < 4; ++t) {
            int col = wc0 + t * 16 + rsub;
            h8 b0 = *(const h8*)(&Bs[col * BKP + kbase]);       // K = kbase..+7
            h8 b1 = *(const h8*)(&Bs[col * BKP + kbase + 8]);   // K = kbase+8..+15
#pragma unroll
            for (int q = 0; q < 8; ++q) { bfr[t][q] = b0[q]; bfr[t][q + 8] = b1[q]; }
        }

        // ---- 2x4 WMMA ----
#pragma unroll
        for (int i = 0; i < 2; ++i)
#pragma unroll
            for (int j = 0; j < 4; ++j)
                acc[i][j] = __builtin_amdgcn_wmma_f32_16x16x32_f16(
                    false, afr[i], false, bfr[j],
                    (short)0, acc[i][j], false, false);
        __syncthreads();
    }

    // ---- epilogue: SiLU (sigmoid via v_rcp_f32), store f16 ----
#pragma unroll
    for (int i = 0; i < 2; ++i) {
#pragma unroll
        for (int j = 0; j < 4; ++j) {
#pragma unroll
            for (int v = 0; v < 8; ++v) {
                int row = mrow + wr0 + i * 16 + v + 8 * hi;     // C/D layout §7.12.2
                int col = ncol + wc0 + j * 16 + rsub;
                float xv = acc[i][j][v];
                float sv = xv * __builtin_amdgcn_rcpf(1.0f + __expf(-xv));
                C[(size_t)row * Nn + col] = (_Float16)sv;
            }
        }
    }
}

// ---------------------------------------------------------------------------
// K4: head GEMV (h2 @ W3) + psl, scaled 1/128, atomic add into per-batch energy
// ---------------------------------------------------------------------------
__global__ void head_kernel(const _Float16* __restrict__ h2,
                            const float* __restrict__ W3,
                            const float* __restrict__ psl,
                            const int*   __restrict__ batch,
                            float* __restrict__ out) {
    int atom = blockIdx.x;
    int tid  = threadIdx.x;
    __shared__ float red[256];
    float a = 0.0f;
    const _Float16* hrow = h2 + (size_t)atom * HID2;
    for (int t = tid; t < HID2; t += 256)
        a += (float)hrow[t] * W3[t];
    red[tid] = a;
    __syncthreads();
    for (int off = 128; off > 0; off >>= 1) {
        if (tid < off) red[tid] += red[tid + off];
        __syncthreads();
    }
    if (tid == 0) {
        float e = (psl[atom] + red[0]) * (1.0f / 128.0f);   // / AVG_ATOMS
        atomicAdd(&out[batch[atom]], e);
    }
}

// ---------------------------------------------------------------------------
// K5: composition term: out[b] += W_comp[species]  (OUT == 1)
// ---------------------------------------------------------------------------
__global__ void comp_kernel(const int* __restrict__ numbers,
                            const int* __restrict__ batch,
                            const float* __restrict__ Wcomp,
                            float* __restrict__ out) {
    int a = blockIdx.x * blockDim.x + threadIdx.x;
    if (a < NATOMS)
        atomicAdd(&out[batch[a]], Wcomp[numbers[a]]);
}

// ---------------------------------------------------------------------------
// Launch
// ---------------------------------------------------------------------------
extern "C" void kernel_launch(void* const* d_in, const int* in_sizes, int n_in,
                              void* d_out, int out_size, void* d_ws, size_t ws_size,
                              hipStream_t stream) {
    const float* pos     = (const float*)d_in[0];
    const float* cells   = (const float*)d_in[1];
    const int*   numbers = (const int*)  d_in[2];
    const int*   eidx    = (const int*)  d_in[3];
    const float* eoff    = (const float*)d_in[4];
    const int*   batch   = (const int*)  d_in[5];
    const float* gamma   = (const float*)d_in[6];
    const float* beta    = (const float*)d_in[7];
    const float* Wps     = (const float*)d_in[8];
    const float* W1      = (const float*)d_in[9];
    const float* W2      = (const float*)d_in[10];
    const float* W3      = (const float*)d_in[11];
    const float* Wcomp   = (const float*)d_in[12];
    float* out = (float*)d_out;

    // workspace layout
    char* w = (char*)d_ws;
    float*    c    = (float*)w;     w += (size_t)NATOMS * NSPEC * 128 * sizeof(float);   // 16 MB
    _Float16* psn  = (_Float16*)w;  w += (size_t)NATOMS * PSDIM * sizeof(_Float16);      // 64 MB
    _Float16* h1   = (_Float16*)w;  w += (size_t)NATOMS * HID1  * sizeof(_Float16);      // 16 MB
    _Float16* h2   = (_Float16*)w;  w += (size_t)NATOMS * HID2  * sizeof(_Float16);      // 16 MB
    float*    psl  = (float*)w;     w += (size_t)NATOMS * sizeof(float);
    _Float16* W1h  = (_Float16*)w;  w += (size_t)PSDIM * HID1 * sizeof(_Float16);        // 8 MB
    _Float16* W2h  = (_Float16*)w;  w += (size_t)HID1  * HID2 * sizeof(_Float16);        // 2 MB

    // zero segment buffer + output
    {
        int nC = NATOMS * NSPEC * 128;
        zero_f32_kernel<<<(nC + 255) / 256, 256, 0, stream>>>(c, nC);
        zero_f32_kernel<<<1, 256, 0, stream>>>(out, out_size);
    }
    // weight conversion (f32 -> f16) — recomputed every call (deterministic)
    {
        int n1 = PSDIM * HID1, n2 = HID1 * HID2;
        cvt_f32_to_f16_kernel<<<(n1 + 255) / 256, 256, 0, stream>>>(W1, W1h, n1);
        cvt_f32_to_f16_kernel<<<(n2 + 255) / 256, 256, 0, stream>>>(W2, W2h, n2);
    }

    // K1: edges
    edge_kernel<<<NEDGES / 256, 256, 0, stream>>>(pos, cells, numbers, eidx, eoff, batch, c);

    // K2: power spectrum + layernorm + W_ps dot
    ps_layernorm_kernel<<<NATOMS, 256, 0, stream>>>(c, gamma, beta, Wps, psn, psl);

    // K3: MLP via WMMA (SiLU fused)
    gemm_silu_kernel<<<dim3(HID1 / BN, NATOMS / BM), 256, 0, stream>>>(psn, W1h, h1, NATOMS, HID1, PSDIM);
    gemm_silu_kernel<<<dim3(HID2 / BN, NATOMS / BM), 256, 0, stream>>>(h1, W2h, h2, NATOMS, HID2, HID1);

    // K4/K5: head + composition
    head_kernel<<<NATOMS, 256, 0, stream>>>(h2, W3, psl, batch, out);
    comp_kernel<<<NATOMS / 256, 256, 0, stream>>>(numbers, batch, Wcomp, out);
}
